// space_aware_block_45646912422392
// MI455X (gfx1250) — matrix operand order repaired and checked by reading, not compiled, not run on hardware
//
#include <hip/hip_runtime.h>
#include <math.h>

typedef __attribute__((ext_vector_type(16))) _Float16 v16h;
typedef __attribute__((ext_vector_type(8)))  _Float16 v8h;
typedef __attribute__((ext_vector_type(4)))  _Float16 v4h;
typedef __attribute__((ext_vector_type(8)))  float    v8f;

#define BATCH  4
#define CCH    64
#define HW     4096
#define JSTEPS (HW / 32)

// -------- Kernel 1: m[b,i] = mean over channels ----------------------------
__global__ __launch_bounds__(256)
void rank1attn_mean_kernel(const float* __restrict__ x, float* __restrict__ m) {
    int idx = blockIdx.x * 256 + threadIdx.x;       // b*HW + i
    int b = idx >> 12;
    int i = idx & (HW - 1);
    const float* xb = x + (size_t)b * CCH * HW + i;
    float s = 0.f;
#pragma unroll
    for (int c = 0; c < CCH; ++c) s += xb[(size_t)c * HW];
    m[idx] = s * (1.0f / 64.0f);
}

// -------- Kernel 2: per-batch max/min of m ---------------------------------
__global__ __launch_bounds__(256)
void rank1attn_minmax_kernel(const float* __restrict__ m, float* __restrict__ mm) {
    __shared__ float smax[256];
    __shared__ float smin[256];
    int b = blockIdx.x, t = threadIdx.x;
    float mx = -INFINITY, mn = INFINITY;
    for (int i = t; i < HW; i += 256) {
        float v = m[b * HW + i];
        mx = fmaxf(mx, v);
        mn = fminf(mn, v);
    }
    smax[t] = mx; smin[t] = mn;
    __syncthreads();
    for (int s = 128; s > 0; s >>= 1) {
        if (t < s) {
            smax[t] = fmaxf(smax[t], smax[t + s]);
            smin[t] = fminf(smin[t], smin[t + s]);
        }
        __syncthreads();
    }
    if (t == 0) { mm[b * 2 + 0] = smax[0]; mm[b * 2 + 1] = smin[0]; }
}

// -------- Kernel 3: pre-pack x into WMMA A-fragment order (f32 -> f16) -----
// Segment s = ((b*4 + ctile)*JSTEPS + jstep)*32 + lane holds that lane's 16
// A-halves for one 16x32 WMMA step: rows c = 16*ctile + (lane&15),
// k-runs [grp*8 .. +7] and [16+grp*8 .. +7] (grp = lane>>4).
__global__ __launch_bounds__(256)
void rank1attn_prepack_kernel(const float* __restrict__ x, _Float16* __restrict__ ap) {
    int g = blockIdx.x * 256 + threadIdx.x;          // 65536 segments
    int lane  = g & 31;
    int jstep = (g >> 5) & (JSTEPS - 1);
    int ct    = (g >> 12) & 3;
    int b     = g >> 14;
    int grp = lane >> 4, l16 = lane & 15;
    int c = ct * 16 + l16;

    const float* src = x + ((size_t)b * CCH + c) * HW + jstep * 32 + grp * 8;
    float4 lo0 = *(const float4*)(src);
    float4 lo1 = *(const float4*)(src + 4);
    float4 hi0 = *(const float4*)(src + 16);
    float4 hi1 = *(const float4*)(src + 20);

    v8h s0, s1;
    s0[0] = (_Float16)lo0.x; s0[1] = (_Float16)lo0.y;
    s0[2] = (_Float16)lo0.z; s0[3] = (_Float16)lo0.w;
    s0[4] = (_Float16)lo1.x; s0[5] = (_Float16)lo1.y;
    s0[6] = (_Float16)lo1.z; s0[7] = (_Float16)lo1.w;
    s1[0] = (_Float16)hi0.x; s1[1] = (_Float16)hi0.y;
    s1[2] = (_Float16)hi0.z; s1[3] = (_Float16)hi0.w;
    s1[4] = (_Float16)hi1.x; s1[5] = (_Float16)hi1.y;
    s1[6] = (_Float16)hi1.z; s1[7] = (_Float16)hi1.w;

    v8h* dst = (v8h*)ap + (size_t)g * 2;
    dst[0] = s0;
    dst[1] = s1;
}

// -------- Kernel 4: fused exp-tile + WMMA GEMM + softmax normalization -----
// grid = (4 batches, 256 i-tiles), block = 128 threads (4 waves).
// Wave w owns c-rows [16w,16w+16); A-fragments stream from prepacked global;
// only the exp tile lives in LDS (double-buffered -> one barrier per step).
__global__ __launch_bounds__(128)
void rank1attn_fused_wmma_kernel(const _Float16* __restrict__ ap,
                                 const float* __restrict__ m,
                                 const float* __restrict__ mm,
                                 float* __restrict__ out) {
    // Row pad 36 halfs = 72 B = 18 banks: 18*l mod 64 distinct for l=0..15
    // (conflict-free B-column reads) and rows stay 8-byte aligned for b64 stores.
    __shared__ _Float16 es[2][16][36];
    __shared__ float    zred[16][8];
    __shared__ float    zfin[16];

    const int b    = blockIdx.x;
    const int i0   = blockIdx.y * 16;
    const int t    = threadIdx.x;
    const int lane = t & 31;
    const int wave = t >> 5;          // c-tile id 0..3
    const int grp  = lane >> 4;       // 0/1 half-wave
    const int l16  = lane & 15;

    // exp-tile split: 16 i's x 8 j-groups of 4
    const int ei  = t >> 3;
    const int ejg = t & 7;

    const float maxm = mm[b * 2 + 0];
    const float minm = mm[b * 2 + 1];
    const float mi   = m[b * HW + i0 + ei];
    const float m2i  = mi * 1.4426950408889634f;          // log2 domain
    const float r2i  = fmaxf(m2i * maxm, m2i * minm);     // exact row max
    float zpart = 0.f;

    const float* mb  = m + (size_t)b * HW;
    const v8h*   apw = (const v8h*)ap +
                       ((size_t)(b * 4 + wave) * JSTEPS * 32 + lane) * 2;

    v8f acc = {0.f, 0.f, 0.f, 0.f, 0.f, 0.f, 0.f, 0.f};

    for (int s = 0; s < JSTEPS; ++s) {
        const int p  = s & 1;
        const int j0 = s * 32;

        // A fragment: two coalesced b128 loads, no LDS
        v8h a0 = apw[0];
        v8h a1 = apw[1];
        apw += 64;

        // exp tile: 4 values per thread, one fma + v_exp_f32 each
        const float4 mj = *(const float4*)(mb + j0 + ejg * 4);
        float e0 = __builtin_amdgcn_exp2f(fmaf(m2i, mj.x, -r2i));
        float e1 = __builtin_amdgcn_exp2f(fmaf(m2i, mj.y, -r2i));
        float e2 = __builtin_amdgcn_exp2f(fmaf(m2i, mj.z, -r2i));
        float e3 = __builtin_amdgcn_exp2f(fmaf(m2i, mj.w, -r2i));
        zpart += (e0 + e1) + (e2 + e3);
        v4h ev;
        ev[0] = (_Float16)e0; ev[1] = (_Float16)e1;
        ev[2] = (_Float16)e2; ev[3] = (_Float16)e3;
        *(v4h*)&es[p][ei][ejg * 4] = ev;    // single ds_store_b64

        __syncthreads();

        // fragments (A straight from global registers, B from LDS columns)
        v16h af, bf;
#pragma unroll
        for (int h = 0; h < 8; ++h) { af[h] = a0[h]; af[8 + h] = a1[h]; }
#pragma unroll
        for (int v = 0; v < 8; ++v) {
            int kb = 2 * (v & 3) + grp * 8 + (v >> 2) * 16;  // B 32x16 layout
            bf[2 * v]     = es[p][l16][kb];
            bf[2 * v + 1] = es[p][l16][kb + 1];
        }
        acc = __builtin_amdgcn_wmma_f32_16x16x32_f16(
            /*neg_a=*/false, af, /*neg_b=*/false, bf,
            /*c_mod=*/(short)0, acc, /*reuse_a=*/false, /*reuse_b=*/false);
    }

    // reduce Z_i (8 partials per i), one reciprocal per i
    zred[ei][ejg] = zpart;
    __syncthreads();
    if (t < 16) {
        float z = 0.f;
#pragma unroll
        for (int q = 0; q < 8; ++q) z += zred[t][q];
        zfin[t] = 1.0f / z;
    }
    __syncthreads();

    // write out: C/D layout -> c = 16*wave + v + 8*grp, i = i0 + l16
    const float rz = zfin[l16];
    float* ob = out + (size_t)b * CCH * HW;
#pragma unroll
    for (int v = 0; v < 8; ++v) {
        int c = wave * 16 + v + 8 * grp;
        ob[(size_t)c * HW + i0 + l16] = acc[v] * rz;
    }
}

extern "C" void kernel_launch(void* const* d_in, const int* in_sizes, int n_in,
                              void* d_out, int out_size, void* d_ws, size_t ws_size,
                              hipStream_t stream) {
    const float* x   = (const float*)d_in[0];
    float*       out = (float*)d_out;

    // workspace: m [4*4096] f32 | mm [8] f32 | apack [2 MB] f16 (16B aligned)
    float*    m  = (float*)d_ws;
    float*    mm = m + BATCH * HW;
    _Float16* ap = (_Float16*)(mm + 8);

    rank1attn_mean_kernel<<<(BATCH * HW) / 256, 256, 0, stream>>>(x, m);
    rank1attn_minmax_kernel<<<BATCH, 256, 0, stream>>>(m, mm);
    rank1attn_prepack_kernel<<<(BATCH * 4 * JSTEPS * 32) / 256, 256, 0, stream>>>(x, ap);

    dim3 grid(BATCH, HW / 16);
    rank1attn_fused_wmma_kernel<<<grid, 128, 0, stream>>>(ap, m, mm, out);
}
